// SCAF_24541443129686
// MI455X (gfx1250) — compile-verified, matrix operand order checked by reading, not run on hardware
//
#include <hip/hip_runtime.h>
#include <hip/hip_bf16.h>

// ---------------------------------------------------------------------------
// LAVT-style cross attention block for MI455X (gfx1250, wave32, WMMA + TDM).
// Big GEMMs: bf16 operands pre-packed once, Tensor Data Mover stages 128x32
// tiles into LDS (padded to 40-elem rows -> conflict-free ds_load_b128),
// double-buffered with s_wait_tensorcnt; V_WMMA_F32_16X16X32_BF16 compute,
// f32 accumulation, fused epilogues. ~200 GFLOP, ~0.8 GB HBM traffic.
// ---------------------------------------------------------------------------

typedef __attribute__((ext_vector_type(16))) __bf16 v16bf;
typedef __attribute__((ext_vector_type(8)))  __bf16 v8bf;
typedef __attribute__((ext_vector_type(8)))  float  v8f;
typedef __attribute__((ext_vector_type(4)))  unsigned int u32x4;
typedef __attribute__((ext_vector_type(8)))  int i32x8;
typedef __attribute__((ext_vector_type(4)))  int i32x4;

#define BQ   16
#define HWQ  4096
#define DQ   512
#define CLQ  768
#define KQ   512
#define NLQ  32
#define HQ   8
#define MROWS (BQ * HWQ)                 // 65536
#define SIMSCALE 0.04419417382415922f    // 512^-0.5
#define EPSQ 1e-5f
#define LDSROW 40                        // 32 data + 8 pad bf16 (80B rows)

__device__ __forceinline__ float gelu_exact(float x) {
    return 0.5f * x * (1.0f + erff(x * 0.7071067811865475f));
}

__device__ __forceinline__ v8f wmma_bf16(v16bf a, v16bf b, v8f c) {
    return __builtin_amdgcn_wmma_f32_16x16x32_bf16(false, a, false, b,
                                                   (short)0, c, false, false);
}

// A/B fragment from f32 row (ISA 7.12.2 lane layout): half=0 -> K {k0..+7,
// k0+16..+23}; half=1 -> +8 on each chunk.
__device__ __forceinline__ v16bf frag_row_f32(const float* __restrict__ row,
                                              int k0, int half) {
    v16bf f;
#pragma unroll
    for (int g = 0; g < 2; ++g) {
        const float* p = row + k0 + half * 8 + g * 16;
#pragma unroll
        for (int j = 0; j < 8; ++j) f[g * 8 + j] = (__bf16)p[j];
    }
    return f;
}

// Same fragment from a 32-elem bf16 LDS row (two 16B ds_load_b128).
__device__ __forceinline__ v16bf frag_row_lds(const __bf16* row, int half) {
    v8bf lo = *(const v8bf*)(row + half * 8);
    v8bf hi = *(const v8bf*)(row + 16 + half * 8);
    v16bf f;
#pragma unroll
    for (int j = 0; j < 8; ++j) { f[j] = lo[j]; f[8 + j] = hi[j]; }
    return f;
}

// ---------------------------------------------------------------------------
// TDM: DMA a (tile_rows x 32) bf16 tile of a 2-D row-major tensor into LDS,
// padding each 64B row with 16B so LDS rows are 80B (bank-conflict free).
// D# per CDNA5 ISA ch.8: group0 {count,lds,global,type=2}; group1 {dsize,
// pad, dims, tile, stride}. Tracked by TENSORcnt.
// ---------------------------------------------------------------------------
__device__ __forceinline__ void tdm_load_tile_bf16(
    const __bf16* gsrc, unsigned lds_off, unsigned kd, unsigned nrows_total,
    unsigned tile_rows)
{
    unsigned long long ga = (unsigned long long)(uintptr_t)gsrc;
    u32x4 g0;
    g0[0] = 1u;                                       // count=1, user D#
    g0[1] = lds_off;                                  // LDS byte address
    g0[2] = (unsigned)(ga & 0xFFFFFFFFu);             // global_addr[31:0]
    g0[3] = (unsigned)((ga >> 32) & 0x1FFFFFFu) | (2u << 30);  // type=2
    i32x8 g1;
    g1[0] = (int)((1u << 16)        // data_size = 2 bytes (bf16)
                | (1u << 20)        // pad_enable
                | (3u << 22)        // pad_interval: every 16 DWORDs (64B)
                | (3u << 25));      // pad_amount: 4 DWORDs (16B)
    g1[1] = (int)((kd & 0xFFFFu) << 16);                       // tensor_dim0 lo
    g1[2] = (int)(((kd >> 16) & 0xFFFFu) | ((nrows_total & 0xFFFFu) << 16));
    g1[3] = (int)(((nrows_total >> 16) & 0xFFFFu) | (32u << 16)); // tile_dim0=32
    g1[4] = (int)(tile_rows & 0xFFFFu);                        // tile_dim1
    g1[5] = (int)kd;                                           // dim0_stride lo
    g1[6] = 0;
    g1[7] = 0;
    i32x4 z4 = {0, 0, 0, 0};
#if defined(__clang_major__) && (__clang_major__ >= 23)
    i32x8 z8 = {0, 0, 0, 0, 0, 0, 0, 0};
    __builtin_amdgcn_tensor_load_to_lds(g0, g1, z4, z4, z8, 0);
#else
    __builtin_amdgcn_tensor_load_to_lds(g0, g1, z4, z4, 0);
#endif
}

// ---------------------------------------------------------------------------
// f32 -> bf16 pack (8 elems/thread, b128 in / b128 out).
// ---------------------------------------------------------------------------
__global__ __launch_bounds__(256) void pack_bf16_k(
    const float* __restrict__ in, __bf16* __restrict__ out, long n8)
{
    long i = (long)blockIdx.x * 256 + threadIdx.x;
    if (i >= n8) return;
    const float4* p = (const float4*)in + i * 2;
    float4 a = p[0], b = p[1];
    v8bf o;
    o[0] = (__bf16)a.x; o[1] = (__bf16)a.y; o[2] = (__bf16)a.z; o[3] = (__bf16)a.w;
    o[4] = (__bf16)b.x; o[5] = (__bf16)b.y; o[6] = (__bf16)b.z; o[7] = (__bf16)b.w;
    ((v8bf*)out)[i] = o;
}

// ---------------------------------------------------------------------------
// Main bf16 GEMM with TDM->LDS double-buffered staging.
// C[m,n] = epi( sum_k A[m,k]*W[n,k] + bias[n] ), A:(M,Kd) bf16, W:(N,Kd) bf16.
// Block 256 thr = 8 waves (4M x 2N); block tile 128x128; wave tile 32x64.
// ---------------------------------------------------------------------------
__global__ __launch_bounds__(256) void gemm_bf16_tdm_k(
    const __bf16* __restrict__ A, const __bf16* __restrict__ W,
    const float* __restrict__ bias, float* __restrict__ C,
    int M, int N, int Kd, int ldc, int doGelu)
{
    __shared__ __bf16 Abuf[2][128 * LDSROW];
    __shared__ __bf16 Bbuf[2][128 * LDSROW];
    const int lane = threadIdx.x & 31;
    const int wave = threadIdx.x >> 5;
    const int half = lane >> 4;
    const int r    = lane & 15;
    const int wm = wave & 3, wn = wave >> 2;
    const int m0 = blockIdx.x * 128;
    const int n0 = blockIdx.y * 128;
    const int nk = Kd / 32;

    v8f acc[2][4];
#pragma unroll
    for (int mt = 0; mt < 2; ++mt)
#pragma unroll
        for (int nt = 0; nt < 4; ++nt)
#pragma unroll
            for (int j = 0; j < 8; ++j) acc[mt][nt][j] = 0.0f;

    if (wave == 0) {   // prologue: DMA tiles for k-step 0 into buffer 0
        tdm_load_tile_bf16(A + (size_t)m0 * Kd,
                           (unsigned)(size_t)&Abuf[0][0], Kd, M, 128);
        tdm_load_tile_bf16(W + (size_t)n0 * Kd,
                           (unsigned)(size_t)&Bbuf[0][0], Kd, N, 128);
    }

    for (int i = 0; i < nk; ++i) {
        const int buf = i & 1;
        if (wave == 0) {
            if (i + 1 < nk) {   // prefetch next pair, then wait current pair
                tdm_load_tile_bf16(A + (size_t)m0 * Kd + (i + 1) * 32,
                                   (unsigned)(size_t)&Abuf[buf ^ 1][0], Kd, M, 128);
                tdm_load_tile_bf16(W + (size_t)n0 * Kd + (i + 1) * 32,
                                   (unsigned)(size_t)&Bbuf[buf ^ 1][0], Kd, N, 128);
                __builtin_amdgcn_s_wait_tensorcnt(2);
            } else {
                __builtin_amdgcn_s_wait_tensorcnt(0);
            }
        }
        __syncthreads();   // tiles in buf are visible to all waves

        v16bf af[2], bfr[4];
#pragma unroll
        for (int mt = 0; mt < 2; ++mt)
            af[mt] = frag_row_lds(&Abuf[buf][(wm * 32 + mt * 16 + r) * LDSROW], half);
#pragma unroll
        for (int nt = 0; nt < 4; ++nt)
            bfr[nt] = frag_row_lds(&Bbuf[buf][(wn * 64 + nt * 16 + r) * LDSROW], half);
#pragma unroll
        for (int mt = 0; mt < 2; ++mt)
#pragma unroll
            for (int nt = 0; nt < 4; ++nt)
                acc[mt][nt] = wmma_bf16(af[mt], bfr[nt], acc[mt][nt]);

        __syncthreads();   // done reading buf before wave0 overwrites it
    }

#pragma unroll
    for (int mt = 0; mt < 2; ++mt)
#pragma unroll
        for (int nt = 0; nt < 4; ++nt)
#pragma unroll
            for (int j = 0; j < 8; ++j) {
                int row = m0 + wm * 32 + mt * 16 + half * 8 + j;
                int col = n0 + wn * 64 + nt * 16 + r;
                float v = acc[mt][nt][j] + bias[col];
                if (doGelu) v = gelu_exact(v);
                C[(size_t)row * ldc + col] = v;
            }
}

// ---------------------------------------------------------------------------
// mm GEMM: C = gelu( (vis .* inorm(wout)) @ W^T + bias ), f32 A path with the
// normalization+product fused into the A-fragment conversion.
// ---------------------------------------------------------------------------
__global__ __launch_bounds__(256) void gemm_mulnorm_k(
    const float* __restrict__ A, const float* __restrict__ A2,
    const float* __restrict__ mean2, const float* __restrict__ rstd2,
    const float* __restrict__ W, const float* __restrict__ bias,
    float* __restrict__ C, int Kd, int ldc)
{
    const int lane = threadIdx.x & 31;
    const int wave = threadIdx.x >> 5;
    const int half = lane >> 4;
    const int r    = lane & 15;
    const int wm = wave & 3, wn = wave >> 2;
    const int m0 = blockIdx.x * 128 + wm * 32;
    const int n0 = blockIdx.y * 128 + wn * 64;
    const int b  = m0 >> 12;                 // m / HW

    v8f acc[2][4];
#pragma unroll
    for (int mt = 0; mt < 2; ++mt)
#pragma unroll
        for (int nt = 0; nt < 4; ++nt)
#pragma unroll
            for (int j = 0; j < 8; ++j) acc[mt][nt][j] = 0.0f;

    for (int k0 = 0; k0 < Kd; k0 += 32) {
        v16bf af[2], bfr[4];
#pragma unroll
        for (int mt = 0; mt < 2; ++mt) {
            const int m = m0 + mt * 16 + r;
            const float* row  = A  + (size_t)m * Kd;
            const float* row2 = A2 + (size_t)m * Kd;
            const float* mp = mean2 + (size_t)b * Kd;
            const float* rp = rstd2 + (size_t)b * Kd;
            v16bf f;
#pragma unroll
            for (int g = 0; g < 2; ++g)
#pragma unroll
                for (int j = 0; j < 8; ++j) {
                    int kk = k0 + half * 8 + g * 16 + j;
                    float z = (row2[kk] - mp[kk]) * rp[kk];
                    f[g * 8 + j] = (__bf16)(row[kk] * z);
                }
            af[mt] = f;
        }
#pragma unroll
        for (int nt = 0; nt < 4; ++nt)
            bfr[nt] = frag_row_f32(W + (size_t)(n0 + nt * 16 + r) * Kd, k0, half);
#pragma unroll
        for (int mt = 0; mt < 2; ++mt)
#pragma unroll
            for (int nt = 0; nt < 4; ++nt)
                acc[mt][nt] = wmma_bf16(af[mt], bfr[nt], acc[mt][nt]);
    }

#pragma unroll
    for (int mt = 0; mt < 2; ++mt)
#pragma unroll
        for (int nt = 0; nt < 4; ++nt)
#pragma unroll
            for (int j = 0; j < 8; ++j) {
                int row = m0 + mt * 16 + half * 8 + j;
                int col = n0 + nt * 16 + r;
                C[(size_t)row * ldc + col] = gelu_exact(acc[mt][nt][j] + bias[col]);
            }
}

// ---------------------------------------------------------------------------
// Column stats over HW per (b, channel): mean & rstd (biased var, eps=1e-5).
// ---------------------------------------------------------------------------
__global__ __launch_bounds__(256) void colstats_k(
    const float* __restrict__ X, int C,
    float* __restrict__ mean, float* __restrict__ rstd)
{
    __shared__ float s1[256], s2[256];
    const int b = blockIdx.x;
    const int t = threadIdx.x;
    const int c = blockIdx.y * 64 + (t & 63);
    const int stripe = t >> 6;
    const float* base = X + (size_t)b * HWQ * C + c;
    float s = 0.0f, ss = 0.0f;
    for (int p = stripe; p < HWQ; p += 4) {
        float x = base[(size_t)p * C];
        s += x; ss += x * x;
    }
    s1[t] = s; s2[t] = ss;
    __syncthreads();
    if (stripe == 0) {
        float S  = s1[t] + s1[t + 64] + s1[t + 128] + s1[t + 192];
        float SS = s2[t] + s2[t + 64] + s2[t + 128] + s2[t + 192];
        float m  = S * (1.0f / HWQ);
        float vr = SS * (1.0f / HWQ) - m * m;
        mean[(size_t)b * C + c] = m;
        rstd[(size_t)b * C + c] = rsqrtf(vr + EPSQ);
    }
}

// ---------------------------------------------------------------------------
// sim[b,h,p,n] = (inorm(q)*scale) . k_t + mask bias.  8 waves x (16 x 32).
// ---------------------------------------------------------------------------
__global__ __launch_bounds__(256) void sim_wmma_k(
    const float* __restrict__ q, const float* __restrict__ qmean,
    const float* __restrict__ qrstd, const float* __restrict__ kt,
    const float* __restrict__ lmask, float* __restrict__ sim)
{
    const int lane = threadIdx.x & 31;
    const int wave = threadIdx.x >> 5;
    const int half = lane >> 4;
    const int r    = lane & 15;
    const int bh = blockIdx.y, b = bh >> 3, h = bh & 7;
    const int m0 = blockIdx.x * 128 + wave * 16;

    v8f acc[2];
#pragma unroll
    for (int nt = 0; nt < 2; ++nt)
#pragma unroll
        for (int j = 0; j < 8; ++j) acc[nt][j] = 0.0f;

    const float* mp = qmean + b * KQ + h * 64;
    const float* rp = qrstd + b * KQ + h * 64;
    const float* qrow = q + ((size_t)(b * HWQ + m0 + r)) * KQ + h * 64;

#pragma unroll
    for (int k0 = 0; k0 < 64; k0 += 32) {
        v16bf a;
#pragma unroll
        for (int g = 0; g < 2; ++g)
#pragma unroll
            for (int j = 0; j < 8; ++j) {
                int kk = k0 + half * 8 + g * 16 + j;
                a[g * 8 + j] = (__bf16)((qrow[kk] - mp[kk]) * rp[kk] * SIMSCALE);
            }
#pragma unroll
        for (int nt = 0; nt < 2; ++nt) {
            v16bf bm = frag_row_f32(
                kt + ((size_t)(b * NLQ + nt * 16 + r)) * KQ + h * 64, k0, half);
            acc[nt] = wmma_bf16(a, bm, acc[nt]);
        }
    }
#pragma unroll
    for (int nt = 0; nt < 2; ++nt)
#pragma unroll
        for (int j = 0; j < 8; ++j) {
            int p = m0 + half * 8 + j;
            int n = nt * 16 + r;
            float v = acc[nt][j] + (lmask[b * NLQ + n] - 1.0f) * 10000.0f;
            sim[((size_t)bh * HWQ + p) * NLQ + n] = v;
        }
}

// ---------------------------------------------------------------------------
// attout(bf16)[b,p,h*64+d] = sum_n att[b,h,p,n]*v[b,h*64+d,n].  K=32, 1 step.
// ---------------------------------------------------------------------------
__global__ __launch_bounds__(256) void attv_wmma_k(
    const float* __restrict__ att, const float* __restrict__ vmat,
    __bf16* __restrict__ out)
{
    const int lane = threadIdx.x & 31;
    const int wave = threadIdx.x >> 5;
    const int half = lane >> 4;
    const int r    = lane & 15;
    const int bh = blockIdx.y, b = bh >> 3, h = bh & 7;
    const int m0 = blockIdx.x * 128 + wave * 16;

    v8f acc[4];
#pragma unroll
    for (int nt = 0; nt < 4; ++nt)
#pragma unroll
        for (int j = 0; j < 8; ++j) acc[nt][j] = 0.0f;

    const float* arow = att + ((size_t)bh * HWQ + m0 + r) * NLQ;
    v16bf a = frag_row_f32(arow, 0, half);
#pragma unroll
    for (int nt = 0; nt < 4; ++nt) {
        const float* vrow = vmat + (size_t)(b * KQ + h * 64 + nt * 16 + r) * NLQ;
        v16bf bm = frag_row_f32(vrow, 0, half);
        acc[nt] = wmma_bf16(a, bm, acc[nt]);
    }
#pragma unroll
    for (int nt = 0; nt < 4; ++nt)
#pragma unroll
        for (int j = 0; j < 8; ++j) {
            int p = m0 + half * 8 + j;
            int d = nt * 16 + r;
            out[((size_t)(b * HWQ + p)) * KQ + h * 64 + d] = (__bf16)acc[nt][j];
        }
}

// ---------------------------------------------------------------------------
// Row softmax over NL=32, in place. One thread per row.
// ---------------------------------------------------------------------------
__global__ __launch_bounds__(256) void softmax_rows_k(float* __restrict__ sim)
{
    size_t row = (size_t)blockIdx.x * 256 + threadIdx.x;
    float* p = sim + row * NLQ;
    float v[NLQ];
    float mx = -1e30f;
#pragma unroll
    for (int i = 0; i < NLQ; ++i) { v[i] = p[i]; mx = fmaxf(mx, v[i]); }
    float s = 0.0f;
#pragma unroll
    for (int i = 0; i < NLQ; ++i) { v[i] = expf(v[i] - mx); s += v[i]; }
    float inv = 1.0f / s;
#pragma unroll
    for (int i = 0; i < NLQ; ++i) p[i] = v[i] * inv;
}

// ---------------------------------------------------------------------------
// latt[b,h,n,p] = softmax over p of sim[b,h,p,n]. One WG per (b,h).
// ---------------------------------------------------------------------------
__global__ __launch_bounds__(256) void softmax_latt_k(
    const float* __restrict__ sim, float* __restrict__ latt)
{
    __shared__ float red[256];
    __shared__ float mxs[NLQ], sms[NLQ];
    const int bh = blockIdx.x, t = threadIdx.x;
    const int n = t & 31, seg = t >> 5;
    const float* S = sim + (size_t)bh * HWQ * NLQ;
    float mx = -1e30f;
    for (int p = seg; p < HWQ; p += 8) mx = fmaxf(mx, S[(size_t)p * NLQ + n]);
    red[t] = mx;
    __syncthreads();
    if (seg == 0) {
        float m = red[n];
#pragma unroll
        for (int k = 1; k < 8; ++k) m = fmaxf(m, red[n + 32 * k]);
        mxs[n] = m;
    }
    __syncthreads();
    float m = mxs[n];
    float s = 0.0f;
    for (int p = seg; p < HWQ; p += 8) s += expf(S[(size_t)p * NLQ + n] - m);
    red[t] = s;
    __syncthreads();
    if (seg == 0) {
        float ss = 0.0f;
#pragma unroll
        for (int k = 0; k < 8; ++k) ss += red[n + 32 * k];
        sms[n] = ss;
    }
    __syncthreads();
    float inv = 1.0f / sms[n];
    float* L = latt + (size_t)bh * NLQ * HWQ;
    for (int p = seg; p < HWQ; p += 8)
        L[(size_t)n * HWQ + p] = expf(S[(size_t)p * NLQ + n] - m) * inv;
}

// ---------------------------------------------------------------------------
// lang_out[b,h*96+c,n] = mask * sum_p latt[b,h,n,p]*inorm(lv)[b,h*96+c,p]
// ---------------------------------------------------------------------------
__global__ __launch_bounds__(256) void langout_k(
    const float* __restrict__ latt, const float* __restrict__ lv,
    const float* __restrict__ lvmean, const float* __restrict__ lvrstd,
    const float* __restrict__ lmask, float* __restrict__ lang_out)
{
    __shared__ float lvs[96 * 64];
    __shared__ float las[32 * 64];
    const int bh = blockIdx.x, b = bh >> 3, h = bh & 7;
    const int t = threadIdx.x, n = t & 31, cg = t >> 5;   // cg in 0..7
    float acc[12];
#pragma unroll
    for (int i = 0; i < 12; ++i) acc[i] = 0.0f;
    const float* mp = lvmean + b * CLQ + h * 96;
    const float* rp = lvrstd + b * CLQ + h * 96;

    for (int pc = 0; pc < HWQ; pc += 64) {
        for (int idx = t; idx < 96 * 64; idx += 256) {
            int cc = idx >> 6, pp = idx & 63;
            float x = lv[((size_t)(b * HWQ + pc + pp)) * CLQ + h * 96 + cc];
            lvs[idx] = (x - mp[cc]) * rp[cc];
        }
        for (int idx = t; idx < 32 * 64; idx += 256) {
            int nn = idx >> 6, pp = idx & 63;
            las[idx] = latt[((size_t)bh * NLQ + nn) * HWQ + pc + pp];
        }
        __syncthreads();
        for (int pp = 0; pp < 64; ++pp) {
            float av = las[n * 64 + pp];
#pragma unroll
            for (int i = 0; i < 12; ++i)
                acc[i] += lvs[(cg * 12 + i) * 64 + pp] * av;
        }
        __syncthreads();
    }
    float mv = lmask[b * NLQ + n];
#pragma unroll
    for (int i = 0; i < 12; ++i) {
        int c = cg * 12 + i;
        lang_out[((size_t)b * CLQ + h * 96 + c) * NLQ + n] = acc[i] * mv;
    }
}

// ---------------------------------------------------------------------------
// Language-side 1x1 conv: out[b,o,n] = epi( sum_i w[o,i]*in[b,i,n](*in2) )
// ---------------------------------------------------------------------------
#define LG_NONE  0
#define LG_GELU  1
#define LG_MASK  2
#define LG_INORM 3
__global__ __launch_bounds__(256) void lang_gemm_k(
    const float* __restrict__ in, const float* __restrict__ in2,
    const float* __restrict__ W, const float* __restrict__ bias,
    const float* __restrict__ lmask, float* __restrict__ out,
    int Cin, int Cout, int mode, int transposed)
{
    __shared__ float ins[128 * 32];
    __shared__ float s1[256], s2[256];
    const int b = blockIdx.x;
    const int t = threadIdx.x;
    const int o = blockIdx.y * 64 + (t >> 2);
    const int nq = t & 3;
    const bool hasMul = (in2 != nullptr);
    float acc[8];
#pragma unroll
    for (int j = 0; j < 8; ++j) acc[j] = 0.0f;

    for (int ci0 = 0; ci0 < Cin; ci0 += 128) {
        for (int idx = t; idx < 128 * 32; idx += 256) {
            int ii = idx >> 5, nn = idx & 31;
            size_t gi = ((size_t)b * Cin + ci0 + ii) * NLQ + nn;
            float x = in[gi];
            if (hasMul) x *= in2[gi];
            ins[idx] = x;
        }
        __syncthreads();
        const float* wr = W + (size_t)o * Cin + ci0;
        for (int ii = 0; ii < 128; ++ii) {
            float wv = wr[ii];
#pragma unroll
            for (int j = 0; j < 8; ++j)
                acc[j] += wv * ins[ii * 32 + nq * 8 + j];
        }
        __syncthreads();
    }
    float bv = bias[o];
    float v[8];
#pragma unroll
    for (int j = 0; j < 8; ++j) v[j] = acc[j] + bv;

    if (mode == LG_GELU) {
#pragma unroll
        for (int j = 0; j < 8; ++j) v[j] = gelu_exact(v[j]);
    } else if (mode == LG_MASK) {
#pragma unroll
        for (int j = 0; j < 8; ++j) v[j] *= lmask[b * NLQ + nq * 8 + j];
    } else if (mode == LG_INORM) {
        float s = 0.0f, ss = 0.0f;
#pragma unroll
        for (int j = 0; j < 8; ++j) { s += v[j]; ss += v[j] * v[j]; }
        s1[t] = s; s2[t] = ss;
        __syncthreads();
        int base = t & ~3;
        float S  = s1[base] + s1[base + 1] + s1[base + 2] + s1[base + 3];
        float SS = s2[base] + s2[base + 1] + s2[base + 2] + s2[base + 3];
        float m  = S * (1.0f / NLQ);
        float vr = SS * (1.0f / NLQ) - m * m;
        float rs = rsqrtf(vr + EPSQ);
#pragma unroll
        for (int j = 0; j < 8; ++j) v[j] = (v[j] - m) * rs;
    }
#pragma unroll
    for (int j = 0; j < 8; ++j) {
        int n = nq * 8 + j;
        if (transposed)
            out[(size_t)b * NLQ * Cout + (size_t)n * Cout + o] = v[j];
        else
            out[((size_t)b * Cout + o) * NLQ + n] = v[j];
    }
}

// ---------------------------------------------------------------------------
extern "C" void kernel_launch(void* const* d_in, const int* in_sizes, int n_in,
                              void* d_out, int out_size, void* d_ws, size_t ws_size,
                              hipStream_t stream) {
    (void)in_sizes; (void)n_in; (void)out_size; (void)ws_size;
    const float* x       = (const float*)d_in[0];
    const float* l       = (const float*)d_in[1];
    const float* lmask   = (const float*)d_in[2];
    const float* w_vis   = (const float*)d_in[3];
    const float* b_vis   = (const float*)d_in[4];
    const float* w_langp = (const float*)d_in[5];
    const float* b_langp = (const float*)d_in[6];
    const float* w_q     = (const float*)d_in[7];
    const float* b_q     = (const float*)d_in[8];
    const float* w_k     = (const float*)d_in[9];
    const float* b_k     = (const float*)d_in[10];
    const float* w_v     = (const float*)d_in[11];
    const float* b_v     = (const float*)d_in[12];
    const float* w_lv    = (const float*)d_in[13];
    const float* b_lv    = (const float*)d_in[14];
    const float* w_W     = (const float*)d_in[15];
    const float* b_W     = (const float*)d_in[16];
    const float* w_langW = (const float*)d_in[17];
    const float* b_langW = (const float*)d_in[18];
    const float* w_mm    = (const float*)d_in[19];
    const float* b_mm    = (const float*)d_in[20];
    const float* w_langmm= (const float*)d_in[21];
    const float* b_langmm= (const float*)d_in[22];

    float* out0 = (float*)d_out;                       // (B,HW,V) row-major
    float* out1 = out0 + (size_t)BQ * HWQ * KQ;        // (B,NL,CL)

    float* ws = (float*)d_ws;
    size_t off = 0;
    float* vis    = ws + off; off += (size_t)MROWS * DQ;        // (m,512) f32
    float* q      = ws + off; off += (size_t)MROWS * KQ;        // (m,512) f32
    float* lv     = ws + off; off += (size_t)MROWS * CLQ;       // (m,768) f32
    float* wout   = ws + off; off += (size_t)MROWS * KQ;        // W-gemm out
    float* sim    = ws + off; off += (size_t)BQ * HQ * HWQ * NLQ;
    float* latt   = ws + off; off += (size_t)BQ * HQ * HWQ * NLQ;
    float* lang   = ws + off; off += (size_t)BQ * CLQ * NLQ;
    float* kt     = ws + off; off += (size_t)BQ * NLQ * KQ;     // (b,n,o)
    float* vv     = ws + off; off += (size_t)BQ * KQ * NLQ;     // (b,o,n)
    float* langO  = ws + off; off += (size_t)BQ * CLQ * NLQ;
    float* langWo = ws + off; off += (size_t)BQ * CLQ * NLQ;
    float* qmean  = ws + off; off += (size_t)BQ * KQ;
    float* qrstd  = ws + off; off += (size_t)BQ * KQ;
    float* lvmean = ws + off; off += (size_t)BQ * CLQ;
    float* lvrstd = ws + off; off += (size_t)BQ * CLQ;
    float* wmean  = ws + off; off += (size_t)BQ * KQ;
    float* wrstd  = ws + off; off += (size_t)BQ * KQ;
    // bf16 regions
    __bf16* bb = (__bf16*)(ws + off);
    size_t boff = 0;
    __bf16* xb   = bb + boff; boff += (size_t)MROWS * DQ;       // x in bf16
    __bf16* attb = bb + boff; boff += (size_t)MROWS * KQ;       // att@v out
    __bf16* wvb  = bb + boff; boff += (size_t)DQ * DQ;
    __bf16* wqb  = bb + boff; boff += (size_t)KQ * DQ;
    __bf16* wlvb = bb + boff; boff += (size_t)CLQ * DQ;
    __bf16* wWb  = bb + boff; boff += (size_t)KQ * KQ;

    // ---- one-time bf16 packs (operands reused by WMMA GEMMs) -------------
    pack_bf16_k<<<(MROWS * (size_t)DQ / 8 + 255) / 256, 256, 0, stream>>>(
        x, xb, (long)MROWS * DQ / 8);
    pack_bf16_k<<<(DQ * DQ / 8 + 255) / 256, 256, 0, stream>>>(w_vis, wvb, DQ * DQ / 8);
    pack_bf16_k<<<(KQ * DQ / 8 + 255) / 256, 256, 0, stream>>>(w_q, wqb, KQ * DQ / 8);
    pack_bf16_k<<<(CLQ * DQ / 8 + 255) / 256, 256, 0, stream>>>(w_lv, wlvb, CLQ * DQ / 8);
    pack_bf16_k<<<(KQ * KQ / 8 + 255) / 256, 256, 0, stream>>>(w_W, wWb, KQ * KQ / 8);

    // ---- language-side projections --------------------------------------
    lang_gemm_k<<<dim3(BQ, CLQ / 64), 256, 0, stream>>>(
        l, nullptr, w_langp, b_langp, nullptr, lang, CLQ, CLQ, LG_GELU, 0);
    lang_gemm_k<<<dim3(BQ, KQ / 64), 256, 0, stream>>>(
        l, nullptr, w_k, b_k, lmask, kt, CLQ, KQ, LG_MASK, 1);   // (b,n,o)
    lang_gemm_k<<<dim3(BQ, KQ / 64), 256, 0, stream>>>(
        l, nullptr, w_v, b_v, lmask, vv, CLQ, KQ, LG_MASK, 0);   // (b,o,n)

    // ---- image-side projections: TDM-staged bf16 WMMA GEMMs --------------
    gemm_bf16_tdm_k<<<dim3(MROWS / 128, DQ / 128), 256, 0, stream>>>(
        xb, wvb, b_vis, vis, MROWS, DQ, DQ, DQ, 1);
    gemm_bf16_tdm_k<<<dim3(MROWS / 128, KQ / 128), 256, 0, stream>>>(
        xb, wqb, b_q, q, MROWS, KQ, DQ, KQ, 0);
    gemm_bf16_tdm_k<<<dim3(MROWS / 128, CLQ / 128), 256, 0, stream>>>(
        xb, wlvb, b_lv, lv, MROWS, CLQ, DQ, CLQ, 0);

    // ---- instance-norm stats (folded into consumers) ---------------------
    colstats_k<<<dim3(BQ, KQ / 64), 256, 0, stream>>>(q, KQ, qmean, qrstd);
    colstats_k<<<dim3(BQ, CLQ / 64), 256, 0, stream>>>(lv, CLQ, lvmean, lvrstd);

    // ---- attention -------------------------------------------------------
    sim_wmma_k<<<dim3(HWQ / 128, BQ * HQ), 256, 0, stream>>>(
        q, qmean, qrstd, kt, lmask, sim);
    softmax_latt_k<<<BQ * HQ, 256, 0, stream>>>(sim, latt);
    softmax_rows_k<<<(BQ * HQ * HWQ) / 256, 256, 0, stream>>>(sim);  // in place
    attv_wmma_k<<<dim3(HWQ / 128, BQ * HQ), 256, 0, stream>>>(sim, vv, attb);

    // ---- language output path -------------------------------------------
    langout_k<<<BQ * HQ, 256, 0, stream>>>(latt, lv, lvmean, lvrstd, lmask, langO);
    lang_gemm_k<<<dim3(BQ, CLQ / 64), 256, 0, stream>>>(
        langO, nullptr, w_langW, b_langW, nullptr, langWo, CLQ, CLQ, LG_INORM, 0);

    // ---- vision output path ----------------------------------------------
    gemm_bf16_tdm_k<<<dim3(MROWS / 128, KQ / 128), 256, 0, stream>>>(
        attb, wWb, b_W, wout, MROWS, KQ, KQ, KQ, 0);
    colstats_k<<<dim3(BQ, KQ / 64), 256, 0, stream>>>(wout, KQ, wmean, wrstd);

    // mm = gelu(conv(vis * inorm(wout)))  -> directly into d_out (B,HW,V)
    gemm_mulnorm_k<<<dim3(MROWS / 128, KQ / 128), 256, 0, stream>>>(
        vis, wout, wmean, wrstd, w_mm, b_mm, out0, KQ, KQ);

    // lang_mm = gelu(conv(lang * langWo)) -> d_out part 2, (B,NL,CL)
    lang_gemm_k<<<dim3(BQ, CLQ / 64), 256, 0, stream>>>(
        lang, langWo, w_langmm, b_langmm, nullptr, out1, CLQ, CLQ, LG_GELU, 1);
}
